// get_model_80968723464489
// MI455X (gfx1250) — compile-verified, matrix operand order checked by reading, not run on hardware
//
#include <hip/hip_runtime.h>

// ---------------------------------------------------------------------------
// bf16 helpers + WMMA types (CDNA5 wave32, v_wmma_f32_16x16x32_bf16)
// ---------------------------------------------------------------------------
typedef __bf16 bf16;
typedef __attribute__((ext_vector_type(16))) __bf16 v16bf;
typedef __attribute__((ext_vector_type(8)))  __bf16 v8bf;
typedef __attribute__((ext_vector_type(8)))  float  v8f;

__device__ __forceinline__ bf16 f2bf(float f) {
  unsigned u = __builtin_bit_cast(unsigned, f);
  unsigned r = (u + 0x7FFFu + ((u >> 16) & 1u)) >> 16;   // round-to-nearest-even
  unsigned short s = (unsigned short)r;
  return __builtin_bit_cast(bf16, s);
}
__device__ __forceinline__ float bf2f(bf16 h) {
  unsigned short s = __builtin_bit_cast(unsigned short, h);
  unsigned u = ((unsigned)s) << 16;
  return __builtin_bit_cast(float, u);
}

// ---------------------------------------------------------------------------
// Fused bf16 GEMM: C[m,n] = epi(alpha * sum_k A[m,k]*B[n,k])
//  * A: bf16, row-major, lda (elements) multiple of 32, rows zero-padded to lda
//  * B: bf16, "n-major k-contiguous" (pre-packed weights / K^T / V^T), ldb mult 32
//  * K-loop has NO bounds checks: strides guarantee [k0,k0+32) in-bounds & pad=0
//  * M is a multiple of 16 by construction. N edge: clamp B row, store 0 to pad.
//  * epilogue: +bias[n]; BN g[n]*y/sqrt(1+eps)+beta[n]; +bf16 residual; relu
//  * batching: z -> (z/zdiv, z%zdiv) two-stride offsets
// ---------------------------------------------------------------------------
__global__ __launch_bounds__(32) void wmma_gemm_kernel(
    const bf16* __restrict__ A, const bf16* __restrict__ Bm, bf16* __restrict__ C,
    int M, int N, int K, int lda, int ldb, int ldc,
    long zdiv, long sA1, long sA2, long sB1, long sB2, long sC1, long sC2,
    const float* __restrict__ bias, const float* __restrict__ gamma,
    const float* __restrict__ beta,
    const bf16* __restrict__ res, long sR1, long sR2, int ldr,
    float alpha, int relu)
{
  int lane = threadIdx.x;
  long z = blockIdx.z;
  long zq = z / zdiv, zr = z % zdiv;
  A  += zq * sA1 + zr * sA2;
  Bm += zq * sB1 + zr * sB2;
  C  += zq * sC1 + zr * sC2;
  if (res) res += zq * sR1 + zr * sR2;

  int m0 = blockIdx.y * 16, n0 = blockIdx.x * 16;
  int hf  = lane >> 4;                 // 0: lanes 0-15, 1: lanes 16-31
  int l15 = lane & 15;
  int mrow = m0 + l15;
  int ncol = n0 + l15;
  int nrow = ncol < N ? ncol : (N - 1);   // clamp B row for pad columns

  const bf16* Ap = A  + (long)mrow * lda + hf * 8;
  const bf16* Bp = Bm + (long)nrow * ldb + hf * 8;

  v8f acc = {0.f, 0.f, 0.f, 0.f, 0.f, 0.f, 0.f, 0.f};

  for (int k0 = 0; k0 < K; k0 += 32) {
    v8bf a0 = *(const v8bf*)(Ap + k0);
    v8bf a1 = *(const v8bf*)(Ap + k0 + 16);
    v8bf b0 = *(const v8bf*)(Bp + k0);
    v8bf b1 = *(const v8bf*)(Bp + k0 + 16);
    v16bf a = __builtin_shufflevector(a0, a1, 0,1,2,3,4,5,6,7,8,9,10,11,12,13,14,15);
    v16bf b = __builtin_shufflevector(b0, b1, 0,1,2,3,4,5,6,7,8,9,10,11,12,13,14,15);
    acc = __builtin_amdgcn_wmma_f32_16x16x32_bf16(false, a, false, b,
                                                  (short)0, acc, false, false);
  }

  const float bninv = 0.99999500003749981f;  // 1/sqrt(1+1e-5)
  bool nv = ncol < N;
  float bia = (nv && bias)  ? bias[ncol]  : 0.f;
  float ga  = (nv && gamma) ? gamma[ncol] : 1.f;
  float be  = (nv && gamma) ? beta[ncol]  : 0.f;
#pragma unroll
  for (int r = 0; r < 8; ++r) {
    int m = m0 + r + hf * 8;
    float v = 0.f;
    if (nv) {
      v = acc[r] * alpha + bia;
      if (gamma) v = ga * v * bninv + be;
      if (res)   v += bf2f(res[(long)m * ldr + ncol]);
      if (relu)  v = fmaxf(v, 0.f);
    }
    C[(long)m * ldc + ncol] = f2bf(v);
  }
}

// ---------------------------------------------------------------------------
// Weight pre-packing: fp32 W -> bf16 Wt[n*KP + k], zero K-pad
//   layoutNK=0: W stored (K,N) row-major (transpose); =1: W stored (N,K)
// ---------------------------------------------------------------------------
__global__ void prep_weight_kernel(const float* __restrict__ W, bf16* __restrict__ Wt,
                                   long total, int K, int N, int KP, int layoutNK) {
  long id = (long)blockIdx.x * blockDim.x + threadIdx.x;
  if (id >= total) return;
  int k = (int)(id % KP); long n = id / KP;
  float v = 0.f;
  if (k < K) v = layoutNK ? W[n * (long)K + k] : W[(long)k * N + n];
  Wt[id] = f2bf(v);
}

// fp32 rows -> bf16 rows with stride KP, zero pad
__global__ void pack_rows_kernel(const float* __restrict__ src, bf16* __restrict__ dst,
                                 long total, int C, int sld, int KP) {
  long id = (long)blockIdx.x * blockDim.x + threadIdx.x;
  if (id >= total) return;
  int k = (int)(id % KP); long r = id / KP;
  float v = (k < C) ? src[r * (long)sld + k] : 0.f;
  dst[id] = f2bf(v);
}

// ---------------------------------------------------------------------------
// Point-cloud kernels (xyz kept fp32; features bf16)
// ---------------------------------------------------------------------------
__global__ void transpose_b3n_kernel(const float* __restrict__ in,
                                     float* __restrict__ out, long total, int N) {
  long id = (long)blockIdx.x * blockDim.x + threadIdx.x;
  if (id >= total) return;
  int c = (int)(id % 3); long rem = id / 3;
  int n = (int)(rem % N); long b = rem / N;
  out[id] = in[(b * 3 + c) * (long)N + n];
}

__global__ void fps_kernel(const float* __restrict__ xyz, int N, int npoint,
                           int* __restrict__ out_idx) {
  int b = blockIdx.x;
  const float* p = xyz + (long)b * N * 3;
  __shared__ float dist[4096];
  __shared__ float rv[256];
  __shared__ int   ri[256];
  int t = threadIdx.x;
  for (int i = t; i < N; i += 256) dist[i] = 1e10f;
  __syncthreads();
  int far = 0;
  for (int s = 0; s < npoint; ++s) {
    if (t == 0) out_idx[b * npoint + s] = far;
    float cx = p[far * 3], cy = p[far * 3 + 1], cz = p[far * 3 + 2];
    float best = -1.f; int bi = 0x7FFFFFFF;
    for (int i = t; i < N; i += 256) {
      float dx = p[i * 3] - cx, dy = p[i * 3 + 1] - cy, dz = p[i * 3 + 2] - cz;
      float d = dx * dx + dy * dy + dz * dz;
      float dm = fminf(dist[i], d);
      dist[i] = dm;
      if (dm > best) { best = dm; bi = i; }
    }
    rv[t] = best; ri[t] = bi;
    __syncthreads();
    for (int o = 128; o > 0; o >>= 1) {
      if (t < o) {
        if (rv[t + o] > rv[t] || (rv[t + o] == rv[t] && ri[t + o] < ri[t])) {
          rv[t] = rv[t + o]; ri[t] = ri[t + o];
        }
      }
      __syncthreads();
    }
    far = ri[0];
    __syncthreads();
  }
}

__global__ void gather_xyz_kernel(const float* __restrict__ xyz,
                                  const int* __restrict__ idx,
                                  float* __restrict__ out, long total, int S, int N) {
  long id = (long)blockIdx.x * blockDim.x + threadIdx.x;
  if (id >= total) return;
  long b = id / S;
  int g = idx[id];
  out[id * 3 + 0] = xyz[(b * N + g) * 3L + 0];
  out[id * 3 + 1] = xyz[(b * N + g) * 3L + 1];
  out[id * 3 + 2] = xyz[(b * N + g) * 3L + 2];
}

__global__ void ball_query_kernel(const float* __restrict__ xyz,
                                  const float* __restrict__ nxyz,
                                  int Nin, int S, float r2, int ns,
                                  int* __restrict__ gi, long total) {
  long id = (long)blockIdx.x * blockDim.x + threadIdx.x;
  if (id >= total) return;
  long b = id / S;
  const float* p = xyz + b * (long)Nin * 3;
  float qx = nxyz[id * 3], qy = nxyz[id * 3 + 1], qz = nxyz[id * 3 + 2];
  int* o = gi + id * (long)ns;
  int cnt = 0, first = 0; bool has = false;
  for (int j = 0; j < Nin && cnt < ns; ++j) {
    float dx = p[j * 3] - qx, dy = p[j * 3 + 1] - qy, dz = p[j * 3 + 2] - qz;
    if (dx * dx + dy * dy + dz * dz <= r2) {
      if (!has) { first = j; has = true; }
      o[cnt++] = j;
    }
  }
  for (; cnt < ns; ++cnt) o[cnt] = first;
}

// grouped tensor (bf16, row stride KP, zero pad): [pts[gi] | xyz[gi]-new_xyz]
__global__ void group_kernel(const float* __restrict__ xyz,
                             const bf16* __restrict__ pts, int Cpts, int pld,
                             const float* __restrict__ nxyz,
                             const int* __restrict__ gi,
                             int Nin, int S, int ns,
                             bf16* __restrict__ out, long total, int Cin, int KP) {
  long id = (long)blockIdx.x * blockDim.x + threadIdx.x;
  if (id >= total) return;
  int c = (int)(id % KP); long rem = id / KP;
  int j = (int)(rem % ns); rem /= ns;
  int s = (int)(rem % S); long b = rem / S;
  float v = 0.f;
  if (c < Cin) {
    int g = gi[(b * S + s) * (long)ns + j];
    if (c < Cpts) {
      v = bf2f(pts[(b * Nin + g) * (long)pld + c]);
    } else {
      int cc = c - Cpts;
      v = xyz[(b * Nin + g) * 3L + cc] - nxyz[(b * S + s) * 3L + cc];
    }
  }
  out[id] = f2bf(v);
}

__global__ void maxpool_kernel(const bf16* __restrict__ in, bf16* __restrict__ out,
                               long R, int ns, int C, int in_ld, int out_ld, int coff) {
  long id = (long)blockIdx.x * blockDim.x + threadIdx.x;
  if (id >= R * (long)C) return;
  long r = id / C; int c = (int)(id % C);
  const bf16* p = in + r * (long)ns * in_ld + c;
  float m = bf2f(p[0]);
  for (int j = 1; j < ns; ++j) m = fmaxf(m, bf2f(p[(long)j * in_ld]));
  out[r * (long)out_ld + coff + c] = f2bf(m);
}

__global__ void rowcopy_bf_kernel(const bf16* __restrict__ src, bf16* __restrict__ dst,
                                  long R, int C, int sld, int dld, int soff, int doff) {
  long id = (long)blockIdx.x * blockDim.x + threadIdx.x;
  if (id >= R * (long)C) return;
  long r = id / C; int c = (int)(id % C);
  dst[r * (long)dld + doff + c] = src[r * (long)sld + soff + c];
}

__global__ void interp_weights_kernel(const float* __restrict__ xyz1,
                                      const float* __restrict__ xyz2,
                                      long R, int N1, int S2,
                                      int* __restrict__ idx3,
                                      float* __restrict__ w3) {
  long id = (long)blockIdx.x * blockDim.x + threadIdx.x;
  if (id >= R) return;
  long b = id / N1;
  const float* q  = xyz1 + id * 3;
  const float* p2 = xyz2 + b * (long)S2 * 3;
  float d0 = 3e38f, d1 = 3e38f, d2 = 3e38f;
  int i0 = 0, i1 = 0, i2 = 0;
  for (int j = 0; j < S2; ++j) {
    float dx = q[0] - p2[j * 3], dy = q[1] - p2[j * 3 + 1], dz = q[2] - p2[j * 3 + 2];
    float d = dx * dx + dy * dy + dz * dz;
    if (d < d0)      { d2 = d1; i2 = i1; d1 = d0; i1 = i0; d0 = d; i0 = j; }
    else if (d < d1) { d2 = d1; i2 = i1; d1 = d; i1 = j; }
    else if (d < d2) { d2 = d; i2 = j; }
  }
  float w0 = 1.f / (d0 + 1e-8f), w1 = 1.f / (d1 + 1e-8f), w2 = 1.f / (d2 + 1e-8f);
  float s = w0 + w1 + w2;
  idx3[id * 3] = i0; idx3[id * 3 + 1] = i1; idx3[id * 3 + 2] = i2;
  w3[id * 3] = w0 / s; w3[id * 3 + 1] = w1 / s; w3[id * 3 + 2] = w2 / s;
}

__global__ void interp_apply_kernel(const bf16* __restrict__ pts2,
                                    const int* __restrict__ idx3,
                                    const float* __restrict__ w3,
                                    bf16* __restrict__ dst,
                                    long R, int N1, int S2, int C2, int dld, int doff) {
  long id = (long)blockIdx.x * blockDim.x + threadIdx.x;
  if (id >= R * (long)C2) return;
  long r = id / C2; int c = (int)(id % C2);
  long b = r / N1;
  const int*   ii = idx3 + r * 3;
  const float* ww = w3   + r * 3;
  const bf16* base = pts2 + b * (long)S2 * C2;
  float v = ww[0] * bf2f(base[(long)ii[0] * C2 + c]) +
            ww[1] * bf2f(base[(long)ii[1] * C2 + c]) +
            ww[2] * bf2f(base[(long)ii[2] * C2 + c]);
  dst[r * (long)dld + doff + c] = f2bf(v);
}

__global__ void layernorm_kernel(const bf16* __restrict__ x, bf16* __restrict__ y,
                                 int D, const float* __restrict__ g,
                                 const float* __restrict__ b) {
  long row = blockIdx.x;
  x += row * (long)D; y += row * (long)D;
  __shared__ float s1[256]; __shared__ float s2[256];
  int t = threadIdx.x;
  float a = 0.f, q = 0.f;
  for (int i = t; i < D; i += 256) { float v = bf2f(x[i]); a += v; q += v * v; }
  s1[t] = a; s2[t] = q;
  __syncthreads();
  for (int o = 128; o > 0; o >>= 1) {
    if (t < o) { s1[t] += s1[t + o]; s2[t] += s2[t + o]; }
    __syncthreads();
  }
  float mean = s1[0] / D;
  float var  = s2[0] / D - mean * mean;
  float inv  = rsqrtf(var + 1e-5f);
  for (int i = t; i < D; i += 256)
    y[i] = f2bf((bf2f(x[i]) - mean) * inv * g[i] + b[i]);
}

// softmax over 16 values, row stride 32 (pad untouched = 0)
__global__ void softmax16_kernel(bf16* __restrict__ a, long rows) {
  long id = (long)blockIdx.x * blockDim.x + threadIdx.x;
  if (id >= rows) return;
  bf16* p = a + id * 32;
  float e[16];
  float m = bf2f(p[0]);
  for (int i = 1; i < 16; ++i) m = fmaxf(m, bf2f(p[i]));
  float s = 0.f;
  for (int i = 0; i < 16; ++i) { e[i] = expf(bf2f(p[i]) - m); s += e[i]; }
  float inv = 1.f / s;
  for (int i = 0; i < 16; ++i) p[i] = f2bf(e[i] * inv);
}

// V^T pack: vT[z=(b*4+h)][n=0..255][k=0..31] = qkv[b*16+k][2048+h*256+n], k>=16 -> 0
__global__ void transpose_v_kernel(const bf16* __restrict__ qkv,
                                   bf16* __restrict__ vT, long total) {
  long id = (long)blockIdx.x * blockDim.x + threadIdx.x;
  if (id >= total) return;
  int k = (int)(id % 32); long rem = id / 32;
  int n = (int)(rem % 256); long z = rem / 256;
  long b = z >> 2; int h2 = (int)(z & 3);
  bf16 v = __builtin_bit_cast(bf16, (unsigned short)0);
  if (k < 16) v = qkv[(b * 16 + k) * 3072L + 2048 + h2 * 256 + n];
  vT[id] = v;
}

// logits (bf16, stride 32, cols 0..1) -> log_softmax fp32
__global__ void logsoftmax2_kernel(const bf16* __restrict__ logits,
                                   float* __restrict__ out, long R) {
  long id = (long)blockIdx.x * blockDim.x + threadIdx.x;
  if (id >= R) return;
  float a = bf2f(logits[id * 32]), b = bf2f(logits[id * 32 + 1]);
  float m = fmaxf(a, b);
  float lse = m + logf(expf(a - m) + expf(b - m));
  out[id * 2]     = a - lse;
  out[id * 2 + 1] = b - lse;
}

// l4 (B,16,1024) bf16 -> out (B,1024,16) fp32
__global__ void transpose_l4_kernel(const bf16* __restrict__ l4,
                                    float* __restrict__ out, long total) {
  long id = (long)blockIdx.x * blockDim.x + threadIdx.x;
  if (id >= total) return;
  int s = (int)(id % 16); long rem = id / 16;
  int d = (int)(rem % 1024); long b = rem / 1024;
  out[id] = bf2f(l4[(b * 16 + s) * 1024L + d]);
}

// ---------------------------------------------------------------------------
// Host-side helpers
// ---------------------------------------------------------------------------
static inline unsigned eg(long n) { return (unsigned)((n + 255) / 256); }
static inline int rnd32(int x) { return (x + 31) & ~31; }

struct Layer { const bf16* Wt; const float* bia; const float* bet; const float* gam; int cin, cout; };

static void launch_gemm(hipStream_t st,
                        const bf16* A, const bf16* Bm, bf16* C,
                        int M, int N, int K, int lda, int ldb, int ldc,
                        int batches, long zdiv,
                        long sA1, long sA2, long sB1, long sB2, long sC1, long sC2,
                        const float* bias, const float* gamma, const float* beta,
                        const bf16* res, long sR1, long sR2, int ldr,
                        float alpha, int relu) {
  int NP = rnd32(N);
  dim3 g((unsigned)(NP / 16), (unsigned)((M + 15) / 16), (unsigned)batches);
  wmma_gemm_kernel<<<g, dim3(32), 0, st>>>(A, Bm, C, M, N, K, lda, ldb, ldc,
                                           zdiv, sA1, sA2, sB1, sB2, sC1, sC2,
                                           bias, gamma, beta, res, sR1, sR2, ldr,
                                           alpha, relu);
}

static void run_branch(hipStream_t st, const float* xyz_in, const bf16* pts_bf,
                       int Cpts, int pld, const float* new_xyz, int* gi,
                       int Bn, int Nin, int S, float radius, int ns,
                       const Layer* Ls, int nl,
                       bf16* s0, bf16* s1, bf16* outbuf, int out_ld, int coff) {
  long bs = (long)Bn * S;
  ball_query_kernel<<<eg(bs), 256, 0, st>>>(xyz_in, new_xyz, Nin, S,
                                            radius * radius, ns, gi, bs);
  int Cin = Cpts + 3;
  int KP0 = rnd32(Cin);
  long tot = bs * ns * KP0;
  group_kernel<<<eg(tot), 256, 0, st>>>(xyz_in, pts_bf, Cpts, pld, new_xyz, gi,
                                        Nin, S, ns, s0, tot, Cin, KP0);
  int M = (int)(bs * ns);
  bf16* bufs[2] = { s0, s1 };
  bf16* cur = s0; int pi = 0; int cin = Cin;
  for (int l = 0; l < nl; ++l) {
    bf16* dst = bufs[(pi + 1) & 1];
    launch_gemm(st, cur, Ls[l].Wt, dst, M, Ls[l].cout, cin,
                rnd32(cin), rnd32(cin), rnd32(Ls[l].cout),
                1, 1, 0, 0, 0, 0, 0, 0,
                Ls[l].bia, Ls[l].gam, Ls[l].bet, nullptr, 0, 0, 0, 1.f, 1);
    cur = dst; pi = (pi + 1) & 1; cin = Ls[l].cout;
  }
  maxpool_kernel<<<eg(bs * cin), 256, 0, st>>>(cur, outbuf, bs, ns, cin,
                                               rnd32(cin), out_ld, coff);
}

static void run_fp(hipStream_t st, const float* xyz1, const float* xyz2,
                   const bf16* pts1, int C1, int p1ld, const bf16* pts2, int C2,
                   int Bn, int N1, int S2, const Layer* Ls, int nl,
                   int* iidx, float* iw, bf16* s0, bf16* s1, bf16* outbuf) {
  long R = (long)Bn * N1;
  interp_weights_kernel<<<eg(R), 256, 0, st>>>(xyz1, xyz2, R, N1, S2, iidx, iw);
  int Ct = C1 + C2;    // always a multiple of 32 here
  if (pts1)
    rowcopy_bf_kernel<<<eg(R * C1), 256, 0, st>>>(pts1, s0, R, C1, p1ld, Ct, 0, 0);
  interp_apply_kernel<<<eg(R * C2), 256, 0, st>>>(pts2, iidx, iw, s0, R, N1, S2,
                                                  C2, Ct, C1);
  bf16* bufs[2] = { s0, s1 };
  bf16* cur = s0; int pi = 0; int cin = Ct;
  for (int l = 0; l < nl; ++l) {
    bf16* dst = (l == nl - 1) ? outbuf : bufs[(pi + 1) & 1];
    launch_gemm(st, cur, Ls[l].Wt, dst, (int)R, Ls[l].cout, cin,
                rnd32(cin), rnd32(cin), rnd32(Ls[l].cout),
                1, 1, 0, 0, 0, 0, 0, 0,
                Ls[l].bia, Ls[l].gam, Ls[l].bet, nullptr, 0, 0, 0, 1.f, 1);
    cur = dst; pi = (pi + 1) & 1; cin = Ls[l].cout;
  }
}

// ---------------------------------------------------------------------------
// Parameter layout: d_in[0]=xyz; params in JAX pytree order (sorted dict keys,
// lists in order). convbn leaves: W,b,beta,g. dense leaves: W,b.
// ---------------------------------------------------------------------------
enum {
  IN_XYZ = 0,
  ENC0 = 1,    // +0 W1,+1 W2,+2 Wo,+3 Wqkv,+4 b1,+5 b2,+6 bo,+7 bqkv,
  ENC1 = 13,   // +8 ln1_b,+9 ln1_g,+10 ln2_b,+11 ln2_g
  FP1 = 25, FP2 = 37, FP3 = 45, FP4 = 53,
  HEAD1 = 61, HEAD2 = 65,
  POS1 = 67, POS2 = 69,
  SA1 = 71, SA2 = 95, SA3 = 119, SA4 = 143,
  TMLP1 = 167, TMLP2 = 169,
  N_PARAMS = 171
};

extern "C" void kernel_launch(void* const* d_in, const int* in_sizes, int n_in,
                              void* d_out, int out_size, void* d_ws, size_t ws_size,
                              hipStream_t stream) {
  (void)in_sizes; (void)out_size; (void)ws_size;
  if (n_in < N_PARAMS) return;
  const int Bn = 8, NPTS = 4096;
  hipStream_t st = stream;
  auto P = [&](int i) { return (const float*)d_in[i]; };

  // ---- bump allocator over d_ws ----
  char* wbase = (char*)d_ws; size_t off = 0;
  auto allocF = [&](size_t n) { float* p = (float*)(wbase + off);
                                off += ((n * 4 + 255) / 256) * 256; return p; };
  auto allocI = [&](size_t n) { int* p = (int*)(wbase + off);
                                off += ((n * 4 + 255) / 256) * 256; return p; };
  auto allocB = [&](size_t n) { bf16* p = (bf16*)(wbase + off);
                                off += ((n * 2 + 255) / 256) * 256; return p; };

  // weight packer (stream-ordered before all GEMMs that use it)
  auto prepW = [&](int idx, int K, int N, int layoutNK) -> const bf16* {
    int KP = rnd32(K);
    bf16* wt = allocB((size_t)N * KP);
    prep_weight_kernel<<<eg((long)N * KP), 256, 0, st>>>(P(idx), wt,
                                                         (long)N * KP, K, N, KP, layoutNK);
    return wt;
  };
  auto L = [&](int idx, int cin, int cout) -> Layer {
    return Layer{ prepW(idx, cin, cout, 0), P(idx + 1), P(idx + 2), P(idx + 3),
                  cin, cout };
  };

  // ---- fp32 coordinate buffers ----
  float* xyz0 = allocF((size_t)Bn * NPTS * 3);
  float* lx1  = allocF((size_t)Bn * 1024 * 3);
  float* lx2  = allocF((size_t)Bn * 256 * 3);
  float* lx3  = allocF((size_t)Bn * 64 * 3);
  float* lx4  = allocF((size_t)Bn * 16 * 3);
  float* iw   = allocF((size_t)Bn * 4096 * 3);
  int* fpsidx = allocI((size_t)Bn * 1024);
  int* gib    = allocI((size_t)Bn * 1024 * 32);
  int* iidx   = allocI((size_t)Bn * 4096 * 3);

  // ---- bf16 feature buffers (row strides all multiples of 32) ----
  bf16* xyz0bf = allocB((size_t)Bn * NPTS * 3);        // pld 3 (scalar access only)
  bf16* lx4bf  = allocB((size_t)128 * 32);             // pos1 GEMM A, K=3 KP=32
  bf16* lp1  = allocB((size_t)Bn * 1024 * 96);
  bf16* lp2  = allocB((size_t)Bn * 256 * 256);
  bf16* lp3  = allocB((size_t)Bn * 64 * 512);
  bf16* lp4  = allocB((size_t)Bn * 16 * 1024);
  bf16* featb = allocB((size_t)128 * 1024);
  bf16* xb    = allocB((size_t)128 * 1024);
  bf16* l4b   = allocB((size_t)128 * 1024);
  bf16* qkv   = allocB((size_t)128 * 3072);
  bf16* attn  = allocB((size_t)32 * 16 * 32);          // (z,16,KP32)
  bf16* vT    = allocB((size_t)32 * 256 * 32);         // (z,n256,KP32)
  bf16* ob    = allocB((size_t)128 * 1024);
  bf16* tb    = allocB((size_t)128 * 2048);
  bf16* ffnb  = allocB((size_t)128 * 2048);
  bf16* l3b   = allocB((size_t)Bn * 64 * 256);
  bf16* l2b   = allocB((size_t)Bn * 256 * 256);
  bf16* l1b   = allocB((size_t)Bn * 1024 * 128);
  bf16* logitsb = allocB((size_t)Bn * 4096 * 32);      // N=2 -> NP=32
  bf16* s0  = allocB((size_t)16777216);                // 32 MB scratch each
  bf16* s1b = allocB((size_t)16777216);

  // ---- input transpose + bf16 copy ----
  transpose_b3n_kernel<<<eg((long)Bn * NPTS * 3), 256, 0, st>>>(
      P(IN_XYZ), xyz0, (long)Bn * NPTS * 3, NPTS);
  pack_rows_kernel<<<eg((long)Bn * NPTS * 3), 256, 0, st>>>(
      xyz0, xyz0bf, (long)Bn * NPTS * 3, 3, 3, 3);

  // ---- SA1: 4096 -> 1024 ----
  fps_kernel<<<Bn, 256, 0, st>>>(xyz0, NPTS, 1024, fpsidx);
  gather_xyz_kernel<<<eg((long)Bn * 1024), 256, 0, st>>>(xyz0, fpsidx, lx1,
                                                         (long)Bn * 1024, 1024, NPTS);
  {
    Layer b0[3] = { L(SA1 + 0, 6, 16),   L(SA1 + 4, 16, 16),  L(SA1 + 8, 16, 32)  };
    Layer b1[3] = { L(SA1 + 12, 6, 32),  L(SA1 + 16, 32, 32), L(SA1 + 20, 32, 64) };
    run_branch(st, xyz0, xyz0bf, 3, 3, lx1, gib, Bn, NPTS, 1024, 0.05f, 16, b0, 3,
               s0, s1b, lp1, 96, 0);
    run_branch(st, xyz0, xyz0bf, 3, 3, lx1, gib, Bn, NPTS, 1024, 0.10f, 32, b1, 3,
               s0, s1b, lp1, 96, 32);
  }

  // ---- SA2: 1024 -> 256 ----
  fps_kernel<<<Bn, 256, 0, st>>>(lx1, 1024, 256, fpsidx);
  gather_xyz_kernel<<<eg((long)Bn * 256), 256, 0, st>>>(lx1, fpsidx, lx2,
                                                        (long)Bn * 256, 256, 1024);
  {
    Layer b0[3] = { L(SA2 + 0, 99, 64),  L(SA2 + 4, 64, 64),  L(SA2 + 8, 64, 128)  };
    Layer b1[3] = { L(SA2 + 12, 99, 64), L(SA2 + 16, 64, 96), L(SA2 + 20, 96, 128) };
    run_branch(st, lx1, lp1, 96, 96, lx2, gib, Bn, 1024, 256, 0.10f, 16, b0, 3,
               s0, s1b, lp2, 256, 0);
    run_branch(st, lx1, lp1, 96, 96, lx2, gib, Bn, 1024, 256, 0.20f, 32, b1, 3,
               s0, s1b, lp2, 256, 128);
  }

  // ---- SA3: 256 -> 64 ----
  fps_kernel<<<Bn, 256, 0, st>>>(lx2, 256, 64, fpsidx);
  gather_xyz_kernel<<<eg((long)Bn * 64), 256, 0, st>>>(lx2, fpsidx, lx3,
                                                       (long)Bn * 64, 64, 256);
  {
    Layer b0[3] = { L(SA3 + 0, 259, 128),  L(SA3 + 4, 128, 196),  L(SA3 + 8, 196, 256)  };
    Layer b1[3] = { L(SA3 + 12, 259, 128), L(SA3 + 16, 128, 196), L(SA3 + 20, 196, 256) };
    run_branch(st, lx2, lp2, 256, 256, lx3, gib, Bn, 256, 64, 0.20f, 16, b0, 3,
               s0, s1b, lp3, 512, 0);
    run_branch(st, lx2, lp2, 256, 256, lx3, gib, Bn, 256, 64, 0.40f, 32, b1, 3,
               s0, s1b, lp3, 512, 256);
  }

  // ---- SA4: 64 -> 16 ----
  fps_kernel<<<Bn, 256, 0, st>>>(lx3, 64, 16, fpsidx);
  gather_xyz_kernel<<<eg((long)Bn * 16), 256, 0, st>>>(lx3, fpsidx, lx4,
                                                       (long)Bn * 16, 16, 64);
  {
    Layer b0[3] = { L(SA4 + 0, 515, 256),  L(SA4 + 4, 256, 256),  L(SA4 + 8, 256, 512)  };
    Layer b1[3] = { L(SA4 + 12, 515, 256), L(SA4 + 16, 256, 384), L(SA4 + 20, 384, 512) };
    run_branch(st, lx3, lp3, 512, 512, lx4, gib, Bn, 64, 16, 0.40f, 16, b0, 3,
               s0, s1b, lp4, 1024, 0);
    run_branch(st, lx3, lp3, 512, 512, lx4, gib, Bn, 64, 16, 0.80f, 32, b1, 3,
               s0, s1b, lp4, 1024, 512);
  }

  // ---- positional encoding, feat = lp4 + pos ----
  pack_rows_kernel<<<eg(128L * 32), 256, 0, st>>>(lx4, lx4bf, 128L * 32, 3, 3, 32);
  const bf16* Wpos1 = prepW(POS1, 3, 1024, 0);
  const bf16* Wpos2 = prepW(POS2, 1024, 1024, 0);
  launch_gemm(st, lx4bf, Wpos1, tb, 128, 1024, 3, 32, 32, 1024,
              1, 1, 0, 0, 0, 0, 0, 0, P(POS1 + 1), nullptr, nullptr,
              nullptr, 0, 0, 0, 1.f, 1);
  launch_gemm(st, tb, Wpos2, featb, 128, 1024, 1024, 1024, 1024, 1024,
              1, 1, 0, 0, 0, 0, 0, 0, P(POS2 + 1), nullptr, nullptr,
              lp4, 0, 0, 1024, 1.f, 0);
  rowcopy_bf_kernel<<<eg(128L * 1024), 256, 0, st>>>(featb, xb, 128, 1024,
                                                     1024, 1024, 0, 0);

  // ---- transformer encoder x2 ----
  for (int e = 0; e < 2; ++e) {
    int pb = (e == 0) ? ENC0 : ENC1;
    const bf16* W1t   = prepW(pb + 0, 1024, 2048, 1);   // (N,K) stored
    const bf16* W2t   = prepW(pb + 1, 2048, 1024, 1);
    const bf16* Wot   = prepW(pb + 2, 1024, 1024, 1);
    const bf16* Wqkvt = prepW(pb + 3, 1024, 3072, 1);
    const float *b1 = P(pb + 4), *b2 = P(pb + 5), *bo = P(pb + 6), *bqkv = P(pb + 7);
    const float *ln1b = P(pb + 8), *ln1g = P(pb + 9), *ln2b = P(pb + 10), *ln2g = P(pb + 11);

    launch_gemm(st, xb, Wqkvt, qkv, 128, 3072, 1024, 1024, 1024, 3072,
                1, 1, 0, 0, 0, 0, 0, 0, bqkv, nullptr, nullptr,
                nullptr, 0, 0, 0, 1.f, 0);
    // scores = QK^T/16 per (b,h); C rows stride 32, pad cols zeroed
    launch_gemm(st, qkv, qkv + 1024, attn, 16, 16, 256, 3072, 3072, 32,
                32, 4, 16L * 3072, 256, 16L * 3072, 256, 2048, 512,
                nullptr, nullptr, nullptr, nullptr, 0, 0, 0, 0.0625f, 0);
    softmax16_kernel<<<eg(512), 256, 0, st>>>(attn, 512);
    transpose_v_kernel<<<eg(32L * 256 * 32), 256, 0, st>>>(qkv, vT, 32L * 256 * 32);
    // o = attn @ V
    launch_gemm(st, attn, vT, ob, 16, 256, 16, 32, 32, 1024,
                32, 4, 2048, 512, 32768, 8192, 16384, 256,
                nullptr, nullptr, nullptr, nullptr, 0, 0, 0, 1.f, 0);
    launch_gemm(st, ob, Wot, tb, 128, 1024, 1024, 1024, 1024, 1024,
                1, 1, 0, 0, 0, 0, 0, 0, bo, nullptr, nullptr,
                xb, 0, 0, 1024, 1.f, 0);
    layernorm_kernel<<<128, 256, 0, st>>>(tb, xb, 1024, ln1g, ln1b);
    launch_gemm(st, xb, W1t, ffnb, 128, 2048, 1024, 1024, 1024, 2048,
                1, 1, 0, 0, 0, 0, 0, 0, b1, nullptr, nullptr,
                nullptr, 0, 0, 0, 1.f, 1);
    launch_gemm(st, ffnb, W2t, tb, 128, 1024, 2048, 2048, 2048, 1024,
                1, 1, 0, 0, 0, 0, 0, 0, b2, nullptr, nullptr,
                xb, 0, 0, 1024, 1.f, 0);
    layernorm_kernel<<<128, 256, 0, st>>>(tb, xb, 1024, ln2g, ln2b);
  }

  // ---- l4 = relu(t@tmlp1+b)@tmlp2+b + feat ----
  const bf16* Wt1 = prepW(TMLP1, 1024, 1024, 0);
  const bf16* Wt2 = prepW(TMLP2, 1024, 1024, 0);
  launch_gemm(st, xb, Wt1, tb, 128, 1024, 1024, 1024, 1024, 1024,
              1, 1, 0, 0, 0, 0, 0, 0, P(TMLP1 + 1), nullptr, nullptr,
              nullptr, 0, 0, 0, 1.f, 1);
  launch_gemm(st, tb, Wt2, l4b, 128, 1024, 1024, 1024, 1024, 1024,
              1, 1, 0, 0, 0, 0, 0, 0, P(TMLP2 + 1), nullptr, nullptr,
              featb, 0, 0, 1024, 1.f, 0);

  // ---- FP decoder ----
  { Layer Ls[2] = { L(FP4 + 0, 1536, 256), L(FP4 + 4, 256, 256) };
    run_fp(st, lx3, lx4, lp3, 512, 512, l4b, 1024, Bn, 64, 16, Ls, 2,
           iidx, iw, s0, s1b, l3b); }
  { Layer Ls[2] = { L(FP3 + 0, 512, 256), L(FP3 + 4, 256, 256) };
    run_fp(st, lx2, lx3, lp2, 256, 256, l3b, 256, Bn, 256, 64, Ls, 2,
           iidx, iw, s0, s1b, l2b); }
  { Layer Ls[2] = { L(FP2 + 0, 352, 256), L(FP2 + 4, 256, 128) };
    run_fp(st, lx1, lx2, lp1, 96, 96, l2b, 256, Bn, 1024, 256, Ls, 2,
           iidx, iw, s0, s1b, l1b); }
  { Layer Ls[3] = { L(FP1 + 0, 128, 128), L(FP1 + 4, 128, 128), L(FP1 + 8, 128, 128) };
    run_fp(st, xyz0, lx1, nullptr, 0, 0, l1b, 128, Bn, 4096, 1024, Ls, 3,
           iidx, iw, s0, s1b, s1b); }

  // ---- segmentation head ----
  const bf16* Wh1 = prepW(HEAD1, 128, 128, 0);
  const bf16* Wh2 = prepW(HEAD2, 128, 2, 0);
  launch_gemm(st, s1b, Wh1, s0, Bn * 4096, 128, 128, 128, 128, 128,
              1, 1, 0, 0, 0, 0, 0, 0,
              P(HEAD1 + 1), P(HEAD1 + 3), P(HEAD1 + 2),
              nullptr, 0, 0, 0, 1.f, 1);
  launch_gemm(st, s0, Wh2, logitsb, Bn * 4096, 2, 128, 128, 128, 32,
              1, 1, 0, 0, 0, 0, 0, 0, P(HEAD2 + 1), nullptr, nullptr,
              nullptr, 0, 0, 0, 1.f, 0);

  float* out = (float*)d_out;
  logsoftmax2_kernel<<<eg((long)Bn * 4096), 256, 0, st>>>(logitsb, out, (long)Bn * 4096);
  transpose_l4_kernel<<<eg((long)Bn * 1024 * 16), 256, 0, st>>>(
      l4b, out + (long)Bn * 4096 * 2, (long)Bn * 1024 * 16);
}
